// SchNetInteractionBlock_1864015806483
// MI455X (gfx1250) — compile-verified, matrix operand order checked by reading, not run on hardware
//
#include <hip/hip_runtime.h>
#include <hip/hip_bf16.h>

typedef _Float16 v16h __attribute__((ext_vector_type(16)));
typedef float    v8f  __attribute__((ext_vector_type(8)));

#define LOG2F_ 0.69314718055994530942f

__device__ __forceinline__ float sspf(float v) {
    // softplus(v) - log(2), numerically stable; native-rate transcendentals
    float a = fabsf(v);
    return fmaxf(v, 0.0f) + __logf(1.0f + __expf(-a)) - LOG2F_;
}

// Build an f16 A-fragment (16x32, ISA 7.12.2 layout) from an f32 row-major source.
// lane<16 holds row m=lane, K = {k0+0..7, k0+16..23}; lane>=16 same rows, K = {k0+8..15, k0+24..31}.
__device__ __forceinline__ v16h load_a_frag_f32(const float* __restrict__ src,
                                                int row_stride, int m_base,
                                                int k0, int klimit) {
    int lane = threadIdx.x & 31;
    int m  = m_base + (lane & 15);
    int hi = (lane >> 4) & 1;
    const float* __restrict__ row = src + (size_t)m * row_stride;  // one 64-bit addr per frag
    v16h a;
#pragma unroll
    for (int h = 0; h < 16; ++h) {
        int k = k0 + h + ((h & 8) ? 8 : 0) + (hi ? 8 : 0);
        float v = (k < klimit) ? row[k] : 0.0f;
        a[h] = (_Float16)v;
    }
    return a;
}

// Pre-swizzled B fragments: fragment f = 32 lanes x 16 halves, contiguous.
__device__ __forceinline__ v16h load_b_frag(const _Float16* __restrict__ bbuf, int frag_id) {
    int lane = threadIdx.x & 31;
    return ((const v16h*)bbuf)[frag_id * 32 + lane];
}

// ---------------- weight prep: transpose+pad+fragment-swizzle to f16 ----------------
// Bin : 8 N-tiles x 4 K-frags (K=128)   -> 16384 halves   B[k][n] = W_in[n*128+k]
// Bf  : 8 N-tiles x 1 K-frag  (K=20->32)->  4096 halves   B[k][n] = k<20 ? W_filter[n*20+k] : 0
// Bout: 8 N-tiles x 4 K-frags (K=128)   -> 16384 halves   B[k][n] = W_out[n*128+k]
__global__ void prep_weights(const float* __restrict__ W_in,
                             const float* __restrict__ W_filter,
                             const float* __restrict__ W_out,
                             _Float16* __restrict__ Bin,
                             _Float16* __restrict__ Bf,
                             _Float16* __restrict__ Bout) {
    int id = blockIdx.x * blockDim.x + threadIdx.x;
    if (id < 16384) {
        int h = id & 15, lane = (id >> 4) & 31, frag = id >> 9;
        int kf = frag & 3, t = frag >> 2;
        int hi = (lane >> 4) & 1;
        int n = t * 16 + (lane & 15);
        int k = kf * 32 + h + ((h & 8) ? 8 : 0) + (hi ? 8 : 0);
        Bin[id] = (_Float16)W_in[n * 128 + k];
    } else if (id < 16384 + 4096) {
        int i = id - 16384;
        int h = i & 15, lane = (i >> 4) & 31, t = i >> 9;
        int hi = (lane >> 4) & 1;
        int n = t * 16 + (lane & 15);
        int k = h + ((h & 8) ? 8 : 0) + (hi ? 8 : 0);
        float v = (k < 20) ? W_filter[n * 20 + k] : 0.0f;
        Bf[i] = (_Float16)v;
    } else if (id < 16384 + 4096 + 16384) {
        int i = id - 16384 - 4096;
        int h = i & 15, lane = (i >> 4) & 31, frag = i >> 9;
        int kf = frag & 3, t = frag >> 2;
        int hi = (lane >> 4) & 1;
        int n = t * 16 + (lane & 15);
        int k = kf * 32 + h + ((h & 8) ? 8 : 0) + (hi ? 8 : 0);
        Bout[i] = (_Float16)W_out[n * 128 + k];
    }
}

// ---------------- h = x @ W_in^T + b_in  (store f32 to L2-resident buffer) ----------
__global__ void gemm_h(const float* __restrict__ x,
                       const _Float16* __restrict__ Bin,
                       const float* __restrict__ b_in,
                       float* __restrict__ hbuf, int Mtiles) {
    int lane = threadIdx.x & 31;
    int t = threadIdx.x >> 5;          // wave-in-block = N-tile 0..7
    v16h b0 = load_b_frag(Bin, t * 4 + 0);
    v16h b1 = load_b_frag(Bin, t * 4 + 1);
    v16h b2 = load_b_frag(Bin, t * 4 + 2);
    v16h b3 = load_b_frag(Bin, t * 4 + 3);
    int hi = (lane >> 4) & 1;
    int n  = t * 16 + (lane & 15);
    float bias = b_in[n];
    for (int mt = blockIdx.x; mt < Mtiles; mt += gridDim.x) {
        int m0 = mt * 16;
        v16h a0 = load_a_frag_f32(x, 128, m0,  0, 128);
        v16h a1 = load_a_frag_f32(x, 128, m0, 32, 128);
        v16h a2 = load_a_frag_f32(x, 128, m0, 64, 128);
        v16h a3 = load_a_frag_f32(x, 128, m0, 96, 128);
        v8f c = {};
        c = __builtin_amdgcn_wmma_f32_16x16x32_f16(false, a0, false, b0, (short)0, c, false, false);
        c = __builtin_amdgcn_wmma_f32_16x16x32_f16(false, a1, false, b1, (short)0, c, false, false);
        c = __builtin_amdgcn_wmma_f32_16x16x32_f16(false, a2, false, b2, (short)0, c, false, false);
        c = __builtin_amdgcn_wmma_f32_16x16x32_f16(false, a3, false, b3, (short)0, c, false, false);
        // single row-base pointer; r-th row is an immediate offset (r*512 bytes)
        float* __restrict__ p = hbuf + ((size_t)(m0 + hi * 8) * 128 + n);
#pragma unroll
        for (int r = 0; r < 8; ++r)
            p[r * 128] = c[r] + bias;
    }
}

// ------ fused: Wij = ssp(f_ij@W_f^T + b_f)*rcut ; gather h[idx_j]; scatter-add agg --
__global__ void edge_kernel(const float* __restrict__ f_ij,
                            const int* __restrict__ idx_i,
                            const int* __restrict__ idx_j,
                            const float* __restrict__ rcut,
                            const _Float16* __restrict__ Bf,
                            const float* __restrict__ b_filter,
                            const float* __restrict__ hbuf,
                            float* __restrict__ agg, int Ptiles) {
    int lane = threadIdx.x & 31;
    int gw = blockIdx.x * (blockDim.x >> 5) + (threadIdx.x >> 5);
    int nw = gridDim.x * (blockDim.x >> 5);
    v16h b[8];
    float bf[8];
    int col = lane & 15;
    int hi  = (lane >> 4) & 1;
#pragma unroll
    for (int t = 0; t < 8; ++t) {
        b[t]  = load_b_frag(Bf, t);
        bf[t] = b_filter[t * 16 + col];
    }
    for (int pt = gw; pt < Ptiles; pt += nw) {
        int ebase = pt * 16;
        if (pt + nw < Ptiles)   // stream-prefetch next tile of f_ij (global_prefetch_b8)
            __builtin_prefetch(&f_ij[(size_t)(pt + nw) * 320 + lane * 10], 0, 1);
        // hoist gather/scatter row bases: one 64-bit addr per row per tile
        const float* __restrict__ hrow[8];
        float*       __restrict__ arow[8];
        float rc[8];
#pragma unroll
        for (int r = 0; r < 8; ++r) {
            int e = ebase + r + hi * 8;
            hrow[r] = hbuf + (size_t)idx_j[e] * 128 + col;
            arow[r] = agg  + (size_t)idx_i[e] * 128 + col;
            rc[r]   = rcut[e];
        }
        v16h a = load_a_frag_f32(f_ij, 20, ebase, 0, 20);
#pragma unroll
        for (int t = 0; t < 8; ++t) {
            v8f c = {};
            c = __builtin_amdgcn_wmma_f32_16x16x32_f16(false, a, false, b[t], (short)0, c, false, false);
#pragma unroll
            for (int r = 0; r < 8; ++r) {
                float w = sspf(c[r] + bf[t]) * rc[r];
                float g = hrow[r][t * 16] * w;          // immediate offset t*64B
                atomicAdd(&arow[r][t * 16], g);
            }
        }
    }
}

// ---------------- out = ssp(agg @ W_out^T + b_out) ----------------------------------
__global__ void gemm_out(const float* __restrict__ aggbuf,
                         const _Float16* __restrict__ Bout,
                         const float* __restrict__ b_out,
                         float* __restrict__ out, int Mtiles) {
    int lane = threadIdx.x & 31;
    int t = threadIdx.x >> 5;
    v16h b0 = load_b_frag(Bout, t * 4 + 0);
    v16h b1 = load_b_frag(Bout, t * 4 + 1);
    v16h b2 = load_b_frag(Bout, t * 4 + 2);
    v16h b3 = load_b_frag(Bout, t * 4 + 3);
    int hi = (lane >> 4) & 1;
    int n  = t * 16 + (lane & 15);
    float bias = b_out[n];
    for (int mt = blockIdx.x; mt < Mtiles; mt += gridDim.x) {
        int m0 = mt * 16;
        v16h a0 = load_a_frag_f32(aggbuf, 128, m0,  0, 128);
        v16h a1 = load_a_frag_f32(aggbuf, 128, m0, 32, 128);
        v16h a2 = load_a_frag_f32(aggbuf, 128, m0, 64, 128);
        v16h a3 = load_a_frag_f32(aggbuf, 128, m0, 96, 128);
        v8f c = {};
        c = __builtin_amdgcn_wmma_f32_16x16x32_f16(false, a0, false, b0, (short)0, c, false, false);
        c = __builtin_amdgcn_wmma_f32_16x16x32_f16(false, a1, false, b1, (short)0, c, false, false);
        c = __builtin_amdgcn_wmma_f32_16x16x32_f16(false, a2, false, b2, (short)0, c, false, false);
        c = __builtin_amdgcn_wmma_f32_16x16x32_f16(false, a3, false, b3, (short)0, c, false, false);
        float* __restrict__ p = out + ((size_t)(m0 + hi * 8) * 128 + n);
#pragma unroll
        for (int r = 0; r < 8; ++r)
            p[r * 128] = sspf(c[r] + bias);
    }
}

extern "C" void kernel_launch(void* const* d_in, const int* in_sizes, int n_in,
                              void* d_out, int out_size, void* d_ws, size_t ws_size,
                              hipStream_t stream) {
    const float* x        = (const float*)d_in[0];
    const float* f_ij     = (const float*)d_in[1];
    const int*   idx_i    = (const int*)d_in[2];
    const int*   idx_j    = (const int*)d_in[3];
    const float* rcut     = (const float*)d_in[4];
    const float* W_in     = (const float*)d_in[5];
    const float* b_in     = (const float*)d_in[6];
    const float* W_filter = (const float*)d_in[7];
    const float* b_filter = (const float*)d_in[8];
    const float* W_out    = (const float*)d_in[9];
    const float* b_out    = (const float*)d_in[10];
    float* out = (float*)d_out;

    const int N = in_sizes[0] / 128;   // 50000 atoms
    const int P = in_sizes[2];         // 1.6M edges
    const int Mtiles = N / 16;         // 3125
    const int Ptiles = P / 16;         // 100000

    char* ws = (char*)d_ws;
    float*     agg  = (float*)ws;                                  // N*128 f32
    float*     hbuf = (float*)(ws + (size_t)N * 128 * 4);          // N*128 f32
    _Float16*  Bin  = (_Float16*)(ws + (size_t)N * 128 * 8);       // 16384 halves
    _Float16*  Bf   = Bin + 16384;                                 // 4096 halves
    _Float16*  Bout = Bf + 4096;                                   // 16384 halves

    hipMemsetAsync(agg, 0, (size_t)N * 128 * 4, stream);
    prep_weights<<<144, 256, 0, stream>>>(W_in, W_filter, W_out, Bin, Bf, Bout);
    gemm_h<<<625, 256, 0, stream>>>(x, Bin, b_in, hbuf, Mtiles);
    edge_kernel<<<4096, 256, 0, stream>>>(f_ij, idx_i, idx_j, rcut, Bf, b_filter,
                                          hbuf, agg, Ptiles);
    gemm_out<<<625, 256, 0, stream>>>(agg, Bout, b_out, out, Mtiles);
}
